// MyBnnNet_65833258713526
// MI455X (gfx1250) — 8-bit matrix operand order repaired and checked by reading, compiled, not run on hardware
//
#include <hip/hip_runtime.h>
#include <math.h>


typedef __attribute__((ext_vector_type(8))) int v8i;

union VI8 { v8i v; int i[8]; };

__device__ __forceinline__ v8i vzero8() {
    v8i z = {0, 0, 0, 0, 0, 0, 0, 0};
    return z;
}

__device__ __forceinline__ v8i wmma_iu8(v8i a, v8i b, v8i c) {
    return __builtin_amdgcn_wmma_i32_16x16x64_iu8(true, a, true, b, c, false, false);
}

__device__ __forceinline__ v8i load_a_row64(const signed char* row, int hi) {
    const int off = hi ? 8 : 0;
    union { v8i v; long long l[4]; } u;
    u.l[0] = *(const long long*)(row + off + 0);
    u.l[1] = *(const long long*)(row + off + 16);
    u.l[2] = *(const long long*)(row + off + 32);
    u.l[3] = *(const long long*)(row + off + 48);
    return u.v;
}

__device__ __forceinline__ v8i load_a_two32(const signed char* p0,
                                            const signed char* p1,
                                            int p1_valid, int hi) {
    const int off = hi ? 8 : 0;
    union { v8i v; long long l[4]; } u;
    u.l[0] = *(const long long*)(p0 + off + 0);
    u.l[1] = *(const long long*)(p0 + off + 16);
    u.l[2] = p1_valid ? *(const long long*)(p1 + off + 0)  : 0ll;
    u.l[3] = p1_valid ? *(const long long*)(p1 + off + 16) : 0ll;
    return u.v;
}

__device__ __forceinline__ v8i load_b_tile(const signed char* base, int lane) {
    const int n    = lane & 15;
    const int koff = (lane >= 16) ? 8 : 0;
    union { v8i v; long long l[4]; } u;
    u.l[0] = *(const long long*)(base + n * 64 + koff + 0);
    u.l[1] = *(const long long*)(base + n * 64 + koff + 16);
    u.l[2] = *(const long long*)(base + n * 64 + koff + 32);
    u.l[3] = *(const long long*)(base + n * 64 + koff + 48);
    return u.v;
}

__global__ void k_pack_weights(const float* __restrict__ w, signed char* __restrict__ dst,
                               int Ktrue, int Ntrue, int Npad, int total) {
    int idx = blockIdx.x * blockDim.x + threadIdx.x;
    if (idx >= total) return;
    int kk = idx & 63;
    int n  = (idx >> 6) % Npad;
    int t  = idx / (64 * Npad);
    int K  = t * 64 + kk;
    signed char v = 0;
    if (K < Ktrue && n < Ntrue)
        v = (w[(size_t)K * Ntrue + n] >= 0.0f) ? (signed char)1 : (signed char)-1;
    dst[idx] = v;
}

__global__ void __launch_bounds__(256)
k_conv1_pool_bn_sign(const float* __restrict__ in, const float* __restrict__ w1,
                     const float* __restrict__ m1, const float* __restrict__ v1,
                     const float* __restrict__ b1, signed char* __restrict__ act1) {
    int idx = blockIdx.x * blockDim.x + threadIdx.x;
    int c  = idx & 31;
    int t  = idx >> 5;
    int ow = t % 31; t /= 31;
    int oh = t % 31; t /= 31;
    int b  = t;
    float best = -3.0e38f;
    #pragma unroll
    for (int dy = 0; dy < 2; ++dy)
    #pragma unroll
    for (int dx = 0; dx < 2; ++dx) {
        int y0 = 2 * oh + dy, x0 = 2 * ow + dx;
        float s = 0.0f;
        #pragma unroll 1
        for (int kh = 0; kh < 3; ++kh)
            #pragma unroll
            for (int kw = 0; kw < 3; ++kw)
                #pragma unroll
                for (int ci = 0; ci < 3; ++ci) {
                    float xv = in[(((size_t)b * 64 + (y0 + kh)) * 64 + (x0 + kw)) * 3 + ci];
                    float wv = w1[(((kh * 3 + kw) * 3) + ci) * 32 + c];
                    s += (wv >= 0.0f) ? xv : -xv;
                }
        best = fmaxf(best, s);
    }
    float rs  = rsqrtf(v1[c] + 1e-3f);
    float thr = m1[c] - b1[c] / rs;
    act1[idx] = (best >= thr) ? (signed char)1 : (signed char)-1;
}

__global__ void __launch_bounds__(128)
k_conv2_gemm(const signed char* __restrict__ act1, const signed char* __restrict__ bp2,
             const float* __restrict__ m2, const float* __restrict__ v2,
             const float* __restrict__ b2, signed char* __restrict__ act2) {
    __shared__ signed char lB[5 * 64 * 64];
    {
        const int4* s = (const int4*)bp2;
        int4* d = (int4*)lB;
        for (int i = threadIdx.x; i < (5 * 64 * 64) / 16; i += blockDim.x) d[i] = s[i];
    }
    __syncthreads();

    const int lane = threadIdx.x & 31;
    const int wave = threadIdx.x >> 5;
    const int hi   = lane >> 4;
    const int mt   = blockIdx.x * 4 + wave;
    const int rp = mt * 16 + (lane & 15);
    const int bA = rp / 196;
    const int rA = rp % 196;
    const int ph = rA / 14, pw = rA % 14;

    VI8 mx[4];
    #pragma unroll 1
    for (int pp = 0; pp < 4; ++pp) {
        const int y = 2 * ph + (pp >> 1);
        const int x = 2 * pw + (pp & 1);
        v8i acc0 = vzero8(), acc1 = vzero8(), acc2 = vzero8(), acc3 = vzero8();
        #pragma unroll 1
        for (int t = 0; t < 5; ++t) {
            const int p0 = 2 * t, p1 = p0 + 1;
            const signed char* q0 =
                act1 + (((size_t)bA * 31 + (y + p0 / 3)) * 31 + (x + p0 % 3)) * 32;
            const signed char* q1 = (p1 < 9)
                ? act1 + (((size_t)bA * 31 + (y + p1 / 3)) * 31 + (x + p1 % 3)) * 32
                : q0;
            v8i a = load_a_two32(q0, q1, (p1 < 9) ? 1 : 0, hi);
            acc0 = wmma_iu8(a, load_b_tile(lB + (t * 64 +  0) * 64, lane), acc0);
            acc1 = wmma_iu8(a, load_b_tile(lB + (t * 64 + 16) * 64, lane), acc1);
            acc2 = wmma_iu8(a, load_b_tile(lB + (t * 64 + 32) * 64, lane), acc2);
            acc3 = wmma_iu8(a, load_b_tile(lB + (t * 64 + 48) * 64, lane), acc3);
        }
        v8i accv[4] = {acc0, acc1, acc2, acc3};
        #pragma unroll
        for (int nt = 0; nt < 4; ++nt) {
            VI8 u; u.v = accv[nt];
            if (pp == 0) {
                mx[nt] = u;
            } else {
                #pragma unroll
                for (int j = 0; j < 8; ++j)
                    if (u.i[j] > mx[nt].i[j]) mx[nt].i[j] = u.i[j];
            }
        }
    }
    #pragma unroll 1
    for (int nt = 0; nt < 4; ++nt) {
        const int ch = nt * 16 + (lane & 15);
        const float rs  = rsqrtf(v2[ch] + 1e-3f);
        const float thr = m2[ch] - b2[ch] / rs;
        #pragma unroll
        for (int j = 0; j < 8; ++j) {
            const int rpj = mt * 16 + hi * 8 + j;
            const int bD = rpj / 196;
            const int rD = rpj % 196;
            const int py = rD / 14, px = rD % 14;
            act2[(((size_t)bD * 14 + py) * 14 + px) * 64 + ch] =
                (((float)mx[nt].i[j]) >= thr) ? (signed char)1 : (signed char)-1;
        }
    }
}

__global__ void __launch_bounds__(256)
k_conv3_gemm(const signed char* __restrict__ act2, const signed char* __restrict__ bp3,
             const float* __restrict__ m3, const float* __restrict__ v3,
             const float* __restrict__ b3, signed char* __restrict__ act3) {
    __shared__ signed char lB[9 * 64 * 64];
    {
        const int4* s = (const int4*)bp3;
        int4* d = (int4*)lB;
        for (int i = threadIdx.x; i < (9 * 64 * 64) / 16; i += blockDim.x) d[i] = s[i];
    }
    __syncthreads();

    const int lane = threadIdx.x & 31;
    const int wave = threadIdx.x >> 5;
    const int hi   = lane >> 4;
    const int mt   = blockIdx.x * 8 + wave;
    const int rp = mt * 16 + (lane & 15);
    const int bA = rp / 144;
    const int rA = rp % 144;
    const int oh = rA / 12, ow = rA % 12;

    v8i acc0 = vzero8(), acc1 = vzero8(), acc2 = vzero8(), acc3 = vzero8();
    #pragma unroll 1
    for (int t = 0; t < 9; ++t) {
        const int kh = t / 3, kw = t % 3;
        const signed char* q =
            act2 + (((size_t)bA * 14 + (oh + kh)) * 14 + (ow + kw)) * 64;
        v8i a = load_a_row64(q, hi);
        acc0 = wmma_iu8(a, load_b_tile(lB + (t * 64 +  0) * 64, lane), acc0);
        acc1 = wmma_iu8(a, load_b_tile(lB + (t * 64 + 16) * 64, lane), acc1);
        acc2 = wmma_iu8(a, load_b_tile(lB + (t * 64 + 32) * 64, lane), acc2);
        acc3 = wmma_iu8(a, load_b_tile(lB + (t * 64 + 48) * 64, lane), acc3);
    }
    v8i accv[4] = {acc0, acc1, acc2, acc3};
    #pragma unroll 1
    for (int nt = 0; nt < 4; ++nt) {
        const int ch = nt * 16 + (lane & 15);
        const float rs  = rsqrtf(v3[ch] + 1e-3f);
        const float thr = m3[ch] - b3[ch] / rs;
        VI8 u; u.v = accv[nt];
        #pragma unroll
        for (int j = 0; j < 8; ++j) {
            const int row = mt * 16 + hi * 8 + j;
            act3[(size_t)row * 64 + ch] =
                (((float)u.i[j]) >= thr) ? (signed char)1 : (signed char)-1;
        }
    }
}

__global__ void __launch_bounds__(128)
k_dense1_gemm(const signed char* __restrict__ act3, const signed char* __restrict__ bpd1,
              const float* __restrict__ m4, const float* __restrict__ v4,
              const float* __restrict__ b4, signed char* __restrict__ act4) {
    const int lane = threadIdx.x & 31;
    const int nt   = threadIdx.x >> 5;
    const int hi   = lane >> 4;
    const int mt   = blockIdx.x;
    const signed char* arow = act3 + (size_t)(mt * 16 + (lane & 15)) * 9216;

    v8i acc = vzero8();
    #pragma unroll 1
    for (int t = 0; t < 144; ++t) {
        v8i a = load_a_row64(arow + t * 64, hi);
        const signed char* bbase = bpd1 + ((size_t)t * 64 + nt * 16) * 64;
        v8i bf = load_b_tile(bbase, lane);
        acc = wmma_iu8(a, bf, acc);
        if (t + 1 < 144)
            __builtin_prefetch(bpd1 + ((size_t)(t + 1) * 64 + nt * 16) * 64, 0, 0);
    }
    const int ch = nt * 16 + (lane & 15);
    const float rs  = rsqrtf(v4[ch] + 1e-3f);
    const float thr = m4[ch] - b4[ch] / rs;
    VI8 u; u.v = acc;
    #pragma unroll
    for (int j = 0; j < 8; ++j) {
        const int row = mt * 16 + hi * 8 + j;
        act4[(size_t)row * 64 + ch] =
            (((float)u.i[j]) >= thr) ? (signed char)1 : (signed char)-1;
    }
}

__global__ void __launch_bounds__(32)
k_dense2_softmax(const signed char* __restrict__ act4, const signed char* __restrict__ bpd2,
                 const float* __restrict__ m5, const float* __restrict__ v5,
                 const float* __restrict__ b5, float* __restrict__ out) {
    const int lane = threadIdx.x & 31;
    const int hi   = lane >> 4;
    const int mt   = blockIdx.x;
    const signed char* arow = act4 + (size_t)(mt * 16 + (lane & 15)) * 64;

    v8i a  = load_a_row64(arow, hi);
    v8i bf = load_b_tile(bpd2, lane);
    v8i c  = wmma_iu8(a, bf, vzero8());

    const int n = lane & 15;
    float mu = 0.0f, rs = 1.0f, be = 0.0f;
    if (n < 10) { mu = m5[n]; rs = rsqrtf(v5[n] + 1e-3f); be = b5[n]; }

    VI8 u; u.v = c;
    #pragma unroll 1
    for (int j = 0; j < 8; ++j) {
        float f = (n < 10) ? ((float)u.i[j] - mu) * rs + be : -3.0e38f;
        float mx = f;
        #pragma unroll
        for (int o = 1; o < 16; o <<= 1) mx = fmaxf(mx, __shfl_xor(mx, o, 16));
        float e = (n < 10) ? expf(f - mx) : 0.0f;
        float s = e;
        #pragma unroll
        for (int o = 1; o < 16; o <<= 1) s += __shfl_xor(s, o, 16);
        if (n < 10)
            out[(size_t)(mt * 16 + hi * 8 + j) * 10 + n] = e / s;
    }
}

extern "C" void kernel_launch(void* const* d_in, const int* in_sizes, int n_in,
                              void* d_out, int out_size, void* d_ws, size_t ws_size,
                              hipStream_t stream) {
    const float* inputs = (const float*)d_in[0];
    const float* w1  = (const float*)d_in[1];
    const float* w2  = (const float*)d_in[2];
    const float* w3  = (const float*)d_in[3];
    const float* d1w = (const float*)d_in[4];
    const float* d2w = (const float*)d_in[5];
    const float* bn1m = (const float*)d_in[6],  *bn1v = (const float*)d_in[7],  *bn1b = (const float*)d_in[8];
    const float* bn2m = (const float*)d_in[9],  *bn2v = (const float*)d_in[10], *bn2b = (const float*)d_in[11];
    const float* bn3m = (const float*)d_in[12], *bn3v = (const float*)d_in[13], *bn3b = (const float*)d_in[14];
    const float* bn4m = (const float*)d_in[15], *bn4v = (const float*)d_in[16], *bn4b = (const float*)d_in[17];
    const float* bn5m = (const float*)d_in[18], *bn5v = (const float*)d_in[19], *bn5b = (const float*)d_in[20];
    float* out = (float*)d_out;

    char* ws = (char*)d_ws;
    const size_t OFF_ACT1 = 0;
    const size_t OFF_ACT2 = 7872512;
    const size_t OFF_ACT3 = 11083776;
    const size_t OFF_ACT4 = 13443072;
    const size_t OFF_BP2  = 13459456;
    const size_t OFF_BP3  = 13479936;
    const size_t OFF_BPD1 = 13516800;
    const size_t OFF_BPD2 = 14106624;

    signed char* act1 = (signed char*)(ws + OFF_ACT1);
    signed char* act2 = (signed char*)(ws + OFF_ACT2);
    signed char* act3 = (signed char*)(ws + OFF_ACT3);
    signed char* act4 = (signed char*)(ws + OFF_ACT4);
    signed char* bp2  = (signed char*)(ws + OFF_BP2);
    signed char* bp3  = (signed char*)(ws + OFF_BP3);
    signed char* bpd1 = (signed char*)(ws + OFF_BPD1);
    signed char* bpd2 = (signed char*)(ws + OFF_BPD2);

    k_pack_weights<<<80,   256, 0, stream>>>(w2,  bp2,  288,  64, 64, 20480);
    k_pack_weights<<<144,  256, 0, stream>>>(w3,  bp3,  576,  64, 64, 36864);
    k_pack_weights<<<2304, 256, 0, stream>>>(d1w, bpd1, 9216, 64, 64, 589824);
    k_pack_weights<<<4,    256, 0, stream>>>(d2w, bpd2, 64,   10, 16, 1024);

    k_conv1_pool_bn_sign<<<30752, 256, 0, stream>>>(inputs, w1, bn1m, bn1v, bn1b, act1);
    k_conv2_gemm<<<784, 128, 0, stream>>>(act1, bp2, bn2m, bn2v, bn2b, act2);
    k_conv3_gemm<<<288, 256, 0, stream>>>(act2, bp3, bn3m, bn3v, bn3b, act3);
    k_dense1_gemm<<<16, 128, 0, stream>>>(act3, bpd1, bn4m, bn4v, bn4b, act4);
    k_dense2_softmax<<<16, 32, 0, stream>>>(act4, bpd2, bn5m, bn5v, bn5b, out);

    (void)in_sizes; (void)n_in; (void)out_size; (void)ws_size;
}
